// LAMM_29368986370240
// MI455X (gfx1250) — matrix operand order repaired and checked by reading, not compiled, not run on hardware
//
#include <hip/hip_runtime.h>
#include <math.h>

// ---------------- problem constants ----------------
#define BQ      256      // batch
#define NTOK    16       // 15 tokens padded to 16
#define DIMQ    512
#define HEADSQ  8
#define DHQ     64
#define FFQ     2048
#define NPATCHQ 14
#define NVQ     5023
#define DIN_MAX 2048     // padded gather stride (>= ceil32(3*max_patch))

// ---------------- WMMA types (CDNA5 wave32) ----------------
typedef __bf16       v16bf __attribute__((ext_vector_type(16)));
typedef float        v8f   __attribute__((ext_vector_type(8)));
typedef unsigned int u32x4 __attribute__((ext_vector_type(4)));

__device__ __forceinline__ unsigned f2bf_pack(float lo, float hi) {
  unsigned ul = __float_as_uint(lo);
  unsigned uh = __float_as_uint(hi);
  ul = (ul + 0x7FFFu + ((ul >> 16) & 1u)) >> 16;   // RNE to bf16
  uh = (uh + 0x7FFFu + ((uh >> 16) & 1u)) & 0xFFFF0000u;
  return ul | uh;
}

__device__ __forceinline__ float gelu_tanh(float x) {
  float x3 = x * x * x;
  return 0.5f * x * (1.0f + tanhf(0.7978845608028654f * (x + 0.044715f * x3)));
}

// =====================================================================
// Generic bf16-WMMA GEMM:  C[M,N] = epi( A[M,K] * W[K,N] + bias (+bias2) )
// Block = 128 threads (4 waves). Tile 64x64, K-step 32.
//
// Contract (all call sites obey):
//   - M is a multiple of 64  -> no A row guards
//   - A rows are readable & zero-padded to ceil32(K) -> no A col guards
// W/C/N/K edges remain guarded; the W guard is taken only on edge tiles
// (uniform branch per K-step).
//
// LDS holds bf16 PAIRS (one dword per K-pair) in fragment order so each
// lane builds its WMMA operands with two aligned ds_load_b128.
// mode 0: store   1: store + residual   2: store gelu   3: scatter atomicAdd
// =====================================================================
__global__ __launch_bounds__(128) void wmma_gemm_kernel(
    const float* __restrict__ A, int lda,
    const float* __restrict__ W, int ldw,
    const float* __restrict__ bias, const float* __restrict__ bias2,
    const float* __restrict__ resid, int ldr,
    float* __restrict__ C, int ldc,
    int M, int N, int K, int mode,
    const int* __restrict__ ids, float* __restrict__ scat)
{
  // 20-dword row stride: rows stay 16B aligned, avoids stride-16 bank pattern
  __shared__ __align__(16) unsigned int sA[64][20];  // [row][pairPos 0..15]
  __shared__ __align__(16) unsigned int sB[64][20];  // [col][kPair   0..15]

  const int tid  = threadIdx.x;
  const int wave = tid >> 5;
  const int lane = tid & 31;
  const int g    = lane >> 4;   // lane half-group
  const int hl   = lane & 15;
  const int tm   = blockIdx.y * 64;
  const int tn   = blockIdx.x * 64;

  v8f acc[4];
  const v8f vzero = {0.f,0.f,0.f,0.f,0.f,0.f,0.f,0.f};
#pragma unroll
  for (int i = 0; i < 4; ++i) acc[i] = vzero;

  for (int k0 = 0; k0 < K; k0 += 32) {
    // ---- stage A tile 64x32 -> packed bf16 pairs, fragment-permuted ----
    // pair p of row r holds A[r][2p],A[r][2p+1]; permuted pos so that
    // lane-group g reads pairs {g*4..g*4+3, 8+g*4..8+g*4+3} contiguously.
    // Unguarded by contract (M%64==0, A zero-padded to ceil32(K)).
#pragma unroll
    for (int i = 0; i < 8; ++i) {
      int idx = i * 128 + tid;            // 1024 pairs
      int r = idx >> 4, p = idx & 15;
      const float2 av = *(const float2*)&A[(long)(tm + r) * lda + k0 + 2 * p];
      int pos = (p & 3) | ((p & 8) >> 1) | ((p & 4) << 1);
      sA[r][pos] = f2bf_pack(av.x, av.y);
    }
    // ---- stage W tile 32x64 -> transposed packed pairs sB[col][kPair] ----
    if (tn + 64 <= N && k0 + 32 <= K) {
      // full tile: unguarded coalesced loads
#pragma unroll
      for (int i = 0; i < 8; ++i) {
        int idx = i * 128 + tid;          // 1024 pairs
        int n = idx & 63, kp = idx >> 6;
        const float* wp = &W[(long)(k0 + 2 * kp) * ldw + tn + n];
        sB[n][kp] = f2bf_pack(wp[0], wp[ldw]);
      }
    } else {
#pragma unroll
      for (int i = 0; i < 8; ++i) {
        int idx = i * 128 + tid;
        int n = idx & 63, kp = idx >> 6;
        int gc = tn + n, gr = k0 + 2 * kp;
        float lo = 0.f, hi = 0.f;
        if (gc < N) {
          if (gr < K)     lo = W[(long)gr * ldw + gc];
          if (gr + 1 < K) hi = W[(long)(gr + 1) * ldw + gc];
        }
        sB[n][kp] = f2bf_pack(lo, hi);
      }
    }
    // prefetch next W tile (global_prefetch_b8)
    if (k0 + 32 < K) {
      int pr = k0 + 32 + (tid >> 6);
      int pc = tn + (tid & 63);
      if (pr < K && pc < N) __builtin_prefetch(&W[(long)pr * ldw + pc], 0, 1);
    }
    __syncthreads();

    // ---- A fragment: 2x ds_load_b128 ----
    union { v16bf v; u32x4 q[2]; } af;
    {
      const u32x4* pA = (const u32x4*)&sA[wave * 16 + hl][g * 8];
      af.q[0] = pA[0];
      af.q[1] = pA[1];
    }
    // ---- 4 N-tiles: B fragment via 2x ds_load_b128 each, then WMMA ----
#pragma unroll
    for (int nt = 0; nt < 4; ++nt) {
      union { v16bf v; u32x4 q[2]; } bfm;
      const u32x4* pB = (const u32x4*)&sB[nt * 16 + hl][g * 4];
      bfm.q[0] = pB[0];
      bfm.q[1] = pB[2];
      acc[nt] = __builtin_amdgcn_wmma_f32_16x16x32_bf16(
          false, af.v, false, bfm.v, (short)0, acc[nt], false, false);
    }
    __syncthreads();
  }

  // ---- epilogue (C layout: VGPR r holds M=r / r+8 in lane halves) ----
#pragma unroll
  for (int nt = 0; nt < 4; ++nt) {
#pragma unroll
    for (int r = 0; r < 8; ++r) {
      int m = tm + wave * 16 + g * 8 + r;
      int n = tn + nt * 16 + hl;
      if (n >= N) continue;              // M%64==0: no row guard needed
      float val = acc[nt][r];
      if (bias)  val += bias[n];
      if (bias2) val += bias2[n];
      if (mode == 1) {
        val += resid[(long)m * ldr + n];
        C[(long)m * ldc + n] = val;
      } else if (mode == 2) {
        C[(long)m * ldc + n] = gelu_tanh(val);
      } else if (mode == 3) {
        int vid = ids[n / 3];
        atomicAdd(&scat[((long)m * NVQ + vid) * 3 + (n % 3)], val);
      } else {
        C[(long)m * ldc + n] = val;
      }
    }
  }
}

// =====================================================================
// LayerNorm over DIM=512, one wave per token (wave32 shuffle reductions)
// =====================================================================
__global__ __launch_bounds__(256) void ln_kernel(
    const float* __restrict__ x, const float* __restrict__ gamma,
    const float* __restrict__ beta, float* __restrict__ out, int ntok)
{
  int wave = threadIdx.x >> 5, lane = threadIdx.x & 31;
  int tok = blockIdx.x * 8 + wave;
  if (tok >= ntok) return;
  const float* xr = x + (long)tok * DIMQ;
  float vals[16];
  float s = 0.f;
#pragma unroll
  for (int i = 0; i < 16; ++i) { vals[i] = xr[lane + 32 * i]; s += vals[i]; }
#pragma unroll
  for (int off = 16; off; off >>= 1) s += __shfl_xor(s, off, 32);
  float mean = s * (1.f / DIMQ);
  float var = 0.f;
#pragma unroll
  for (int i = 0; i < 16; ++i) { float d = vals[i] - mean; var += d * d; }
#pragma unroll
  for (int off = 16; off; off >>= 1) var += __shfl_xor(var, off, 32);
  float rstd = rsqrtf(var * (1.f / DIMQ) + 1e-5f);
  float* orow = out + (long)tok * DIMQ;
#pragma unroll
  for (int i = 0; i < 16; ++i) {
    int d = lane + 32 * i;
    orow[d] = (vals[i] - mean) * rstd * gamma[d] + beta[d];
  }
}

// =====================================================================
// Attention: one block per (batch, head). 15 real tokens (key 15 masked).
// =====================================================================
__global__ __launch_bounds__(256) void attn_kernel(
    const float* __restrict__ q, const float* __restrict__ k,
    const float* __restrict__ v, float* __restrict__ o)
{
  __shared__ float qs[16][DHQ];
  __shared__ float ks[15][DHQ];
  __shared__ float vs[15][DHQ];
  __shared__ float sc[16][16];

  int b = blockIdx.x >> 3;
  int h = blockIdx.x & 7;
  int t = threadIdx.x;
  long base = (long)(b * NTOK) * DIMQ + h * DHQ;   // (b, n=0, h, d=0)

  for (int idx = t; idx < NTOK * DHQ; idx += 256) {
    int n = idx >> 6, d = idx & 63;
    long off = base + (long)n * DIMQ + d;
    qs[n][d] = q[off];
    if (n < 15) { ks[n][d] = k[off]; vs[n][d] = v[off]; }
  }
  __syncthreads();

  if (t < 16 * 15) {
    int qi = t / 15, ki = t % 15;
    float dot = 0.f;
#pragma unroll
    for (int d = 0; d < DHQ; ++d) dot += qs[qi][d] * ks[ki][d];
    sc[qi][ki] = dot * 0.125f;   // DH^-0.5
  }
  __syncthreads();

  if (t < 16) {
    float mx = -1e30f;
    for (int ki = 0; ki < 15; ++ki) mx = fmaxf(mx, sc[t][ki]);
    float sum = 0.f;
    for (int ki = 0; ki < 15; ++ki) { float e = __expf(sc[t][ki] - mx); sc[t][ki] = e; sum += e; }
    float inv = 1.f / sum;
    for (int ki = 0; ki < 15; ++ki) sc[t][ki] *= inv;
  }
  __syncthreads();

  for (int idx = t; idx < NTOK * DHQ; idx += 256) {
    int qi = idx >> 6, d = idx & 63;
    float s = 0.f;
#pragma unroll
    for (int ki = 0; ki < 15; ++ki) s += sc[qi][ki] * vs[ki][d];
    o[base + (long)qi * DIMQ + d] = s;
  }
}

// ---------------- small utility kernels ----------------
// Gathers one patch's coordinates and zero-pads each row to a multiple of
// 32 so the GEMM A-side can load unguarded.
__global__ void gather_kernel(const float* __restrict__ x,
                              const int* __restrict__ ids,
                              float* __restrict__ Xg, int din)
{
  int b = blockIdx.x;
  int dpad = (din + 31) & ~31;
  for (int i = threadIdx.x; i < dpad; i += blockDim.x) {
    float v = 0.f;
    if (i < din) {
      int vid = ids[i / 3];
      v = x[((long)b * NVQ + vid) * 3 + (i % 3)];
    }
    Xg[(long)b * DIN_MAX + i] = v;
  }
}

__global__ void init_tok_kernel(const float* __restrict__ id_tok, float* __restrict__ tok)
{
  int b = blockIdx.x;
  for (int d = threadIdx.x; d < DIMQ; d += blockDim.x) {
    tok[((long)b * NTOK + 0)  * DIMQ + d] = id_tok[d];
    tok[((long)b * NTOK + 15) * DIMQ + d] = 0.f;
  }
}

__global__ void init_dec_kernel(const float* __restrict__ xenc,
                                const float* __restrict__ sem,
                                float* __restrict__ tok)
{
  int b = blockIdx.x;
  for (int i = threadIdx.x; i < NTOK * DIMQ; i += blockDim.x) {
    int n = i >> 9, d = i & 511;
    float val;
    if (n == 0)       val = xenc[(long)b * NTOK * DIMQ + d];  // id token
    else if (n <= 14) val = sem[(n - 1) * DIMQ + d];
    else              val = 0.f;
    tok[((long)b * NTOK + n) * DIMQ + d] = val;
  }
}

__global__ void zero_kernel(float* __restrict__ p, long n)
{
  long i = (long)blockIdx.x * blockDim.x + threadIdx.x;
  if (i < n) p[i] = 0.f;
}

// =====================================================================
// Host orchestration
// =====================================================================
static inline void launch_gemm(const float* A, int lda, const float* W, int ldw,
                               const float* bias, const float* bias2,
                               const float* resid, int ldr,
                               float* C, int ldc, int M, int N, int K, int mode,
                               const int* ids, float* scat, hipStream_t s)
{
  dim3 grid((N + 63) / 64, (M + 63) / 64);
  wmma_gemm_kernel<<<grid, 128, 0, s>>>(A, lda, W, ldw, bias, bias2,
                                        resid, ldr, C, ldc, M, N, K, mode, ids, scat);
}

struct LayerBufs {
  float *h, *xmid, *qb, *kb, *vb, *ob, *ff;
};

static void run_layer(const float* const* P, int l,
                      const float* xin, float* xout, const LayerBufs& W,
                      hipStream_t s)
{
  const int M = BQ * NTOK;
  const float* ln1_g = P[0]  + (long)l * DIMQ;
  const float* ln1_b = P[1]  + (long)l * DIMQ;
  const float* Wq    = P[2]  + (long)l * DIMQ * DIMQ;
  const float* bq    = P[3]  + (long)l * DIMQ;
  const float* Wk    = P[4]  + (long)l * DIMQ * DIMQ;
  const float* bk    = P[5]  + (long)l * DIMQ;
  const float* Wv    = P[6]  + (long)l * DIMQ * DIMQ;
  const float* bv    = P[7]  + (long)l * DIMQ;
  const float* Wo    = P[8]  + (long)l * DIMQ * DIMQ;
  const float* bo    = P[9]  + (long)l * DIMQ;
  const float* ln2_g = P[10] + (long)l * DIMQ;
  const float* ln2_b = P[11] + (long)l * DIMQ;
  const float* W1    = P[12] + (long)l * DIMQ * FFQ;
  const float* b1    = P[13] + (long)l * FFQ;
  const float* W2    = P[14] + (long)l * FFQ * DIMQ;
  const float* b2    = P[15] + (long)l * DIMQ;

  ln_kernel<<<M / 8, 256, 0, s>>>(xin, ln1_g, ln1_b, W.h, M);
  launch_gemm(W.h, DIMQ, Wq, DIMQ, bq, nullptr, nullptr, 0, W.qb, DIMQ, M, DIMQ, DIMQ, 0, nullptr, nullptr, s);
  launch_gemm(W.h, DIMQ, Wk, DIMQ, bk, nullptr, nullptr, 0, W.kb, DIMQ, M, DIMQ, DIMQ, 0, nullptr, nullptr, s);
  launch_gemm(W.h, DIMQ, Wv, DIMQ, bv, nullptr, nullptr, 0, W.vb, DIMQ, M, DIMQ, DIMQ, 0, nullptr, nullptr, s);
  attn_kernel<<<BQ * HEADSQ, 256, 0, s>>>(W.qb, W.kb, W.vb, W.ob);
  launch_gemm(W.ob, DIMQ, Wo, DIMQ, bo, nullptr, xin, DIMQ, W.xmid, DIMQ, M, DIMQ, DIMQ, 1, nullptr, nullptr, s);
  ln_kernel<<<M / 8, 256, 0, s>>>(W.xmid, ln2_g, ln2_b, W.h, M);
  launch_gemm(W.h, DIMQ, W1, FFQ, b1, nullptr, nullptr, 0, W.ff, FFQ, M, FFQ, DIMQ, 2, nullptr, nullptr, s);
  launch_gemm(W.ff, FFQ, W2, DIMQ, b2, nullptr, W.xmid, DIMQ, xout, DIMQ, M, DIMQ, FFQ, 1, nullptr, nullptr, s);
}

extern "C" void kernel_launch(void* const* d_in, const int* in_sizes, int n_in,
                              void* d_out, int out_size, void* d_ws, size_t ws_size,
                              hipStream_t stream)
{
  (void)n_in; (void)out_size; (void)ws_size;

  // ---- input index map (setup_inputs dict order, lists flattened) ----
  const float* x = (const float*)d_in[0];
  const int*   fp_ids[NPATCHQ];
  const float* tokW[NPATCHQ];
  const float* tokb[NPATCHQ];
  const float* outW[NPATCHQ];
  const float* outb[NPATCHQ];
  int din[NPATCHQ];
  for (int j = 0; j < NPATCHQ; ++j) {
    fp_ids[j] = (const int*)d_in[1 + j];
    din[j]    = 3 * in_sizes[1 + j];
    tokW[j]   = (const float*)d_in[15 + j];
    tokb[j]   = (const float*)d_in[29 + j];
    outW[j]   = (const float*)d_in[78 + j];
    outb[j]   = (const float*)d_in[92 + j];
  }
  const float* sem_emb    = (const float*)d_in[43];
  const float* id_tok     = (const float*)d_in[44];
  const float* sem_tokens = (const float*)d_in[45];
  const float* enc[16];
  const float* dec[16];
  for (int i = 0; i < 16; ++i) { enc[i] = (const float*)d_in[46 + i]; dec[i] = (const float*)d_in[62 + i]; }
  float* out = (float*)d_out;

  // ---- workspace carve ----
  const long ACT = (long)BQ * NTOK * DIMQ;   // 2,097,152 floats
  float* ws = (float*)d_ws;
  float* Xg  = ws;  ws += (long)NPATCHQ * BQ * DIN_MAX;
  float* x0  = ws;  ws += ACT;   // tokens
  float* x1  = ws;  ws += ACT;   // enc layer 0 out
  float* x2  = ws;  ws += ACT;   // enc layer 1 out
  float* x3  = ws;  ws += ACT;   // dec tokens
  float* x4  = ws;  ws += ACT;   // dec layer 0 out
  float* x5  = ws;  ws += ACT;   // dec layer 1 out
  LayerBufs L;
  L.h    = ws;  ws += ACT;
  L.xmid = ws;  ws += ACT;
  L.qb   = ws;  ws += ACT;
  L.kb   = ws;  ws += ACT;
  L.vb   = ws;  ws += ACT;
  L.ob   = ws;  ws += ACT;
  L.ff   = ws;  ws += (long)BQ * NTOK * FFQ;

  // ---- zero the scatter output (4, B, NV, 3) ----
  const long NOUT = 4L * BQ * NVQ * 3;
  zero_kernel<<<(int)((NOUT + 255) / 256), 256, 0, stream>>>(out, NOUT);

  // ---- tokenize: gather (zero-padded) + per-patch GEMM into tokens rows 1..14 ----
  init_tok_kernel<<<BQ, 256, 0, stream>>>(id_tok, x0);
  for (int j = 0; j < NPATCHQ; ++j) {
    float* Xgj = Xg + (long)j * BQ * DIN_MAX;
    gather_kernel<<<BQ, 256, 0, stream>>>(x, fp_ids[j], Xgj, din[j]);
    // tokens[:, j+1, :] = Xg_j @ tokW_j + tokb_j + sem_emb[j]
    launch_gemm(Xgj, DIN_MAX, tokW[j], DIMQ, tokb[j], sem_emb + (long)j * DIMQ,
                nullptr, 0, x0 + (long)(j + 1) * DIMQ, NTOK * DIMQ,
                BQ, DIMQ, din[j], 0, nullptr, nullptr, stream);
  }

  // ---- encoder ----
  run_layer(enc, 0, x0, x1, L, stream);
  run_layer(enc, 1, x1, x2, L, stream);

  // ---- decoder tokens + decoder ----
  init_dec_kernel<<<BQ, 256, 0, stream>>>(x2, sem_tokens, x3);
  run_layer(dec, 0, x3, x4, L, stream);
  run_layer(dec, 1, x4, x5, L, stream);

  // ---- merge: 4 layer outputs -> scatter-add output slabs ----
  const float* louts[4] = { x1, x2, x4, x5 };
  for (int s = 0; s < 4; ++s) {
    float* slab = out + (long)s * BQ * NVQ * 3;
    for (int j = 0; j < NPATCHQ; ++j) {
      // Y = x_l[:, j+1, :] @ outW_j + outb_j ; scatter-add by fp_ids_j
      launch_gemm(louts[s] + (long)(j + 1) * DIMQ, NTOK * DIMQ,
                  outW[j], din[j], outb[j], nullptr, nullptr, 0,
                  nullptr, 0, BQ, din[j], DIMQ, 3, fp_ids[j], slab, stream);
    }
  }
}
